// UDRNNModel_30468497998213
// MI455X (gfx1250) — matrix operand order repaired and checked by reading, not compiled, not run on hardware
//
#include <hip/hip_runtime.h>
#include <hip/hip_bf16.h>
#include <math.h>

typedef __attribute__((ext_vector_type(16))) _Float16 v16h;
typedef __attribute__((ext_vector_type(8)))  _Float16 v8h;
typedef __attribute__((ext_vector_type(8)))  float    v8f;

#define B_   64
#define S_   256
#define D_   300
#define NW_  8192
#define CL_  16
#define CD_  30
#define CH_  30
#define H_   128
#define T_   17
#define G3_  384   // 3*H
#define K0P  352   // in0 = 330 padded to mult of 32
#define K1P  256
#define N01  768   // 2 dirs * 384
#define BCH  16    // batch rows per GRU block

// ---------------- WMMA fragment loaders (wave32, 16x16x32 f16) --------------
// A 16x32 f16: lane<16 -> row=lane,   K = g*16 + r      (g = h>>3, r = h&7)
//              lane>=16-> row=lane-16,K = g*16 + 8 + r
__device__ inline v16h load_fragA(const _Float16* base, int lda, int kk, int lane) {
    int l15 = lane & 15, lg = lane >> 4;
    const _Float16* p = base + (size_t)l15 * lda + kk + lg * 8;
    v8h lo = *(const v8h*)(p);
    v8h hi = *(const v8h*)(p + 16);
    v16h a;
#pragma unroll
    for (int i = 0; i < 8; ++i) { a[i] = lo[i]; a[8 + i] = hi[i]; }
    return a;
}
// B 32x16 f16 (Bm stored as N rows x K cols): col N = lane&15, K = g*16 + (lane>>4)*8 + r
__device__ inline v16h load_fragB(const _Float16* base, int ldb, int kk, int lane) {
    int l15 = lane & 15, lg = lane >> 4;
    const _Float16* p = base + (size_t)l15 * ldb + kk + lg * 8;
    v8h lo = *(const v8h*)(p);
    v8h hi = *(const v8h*)(p + 16);
    v16h b;
#pragma unroll
    for (int i = 0; i < 8; ++i) { b[i] = lo[i]; b[8 + i] = hi[i]; }
    return b;
}

// ---------------- Generic WMMA GEMM: C[M,N] = A[M,K] * Bm[N,K]^T + bias -----
__global__ void __launch_bounds__(256) wmma_gemm_kernel(
    const _Float16* __restrict__ A, const _Float16* __restrict__ Bm,
    float* __restrict__ C, int M, int N, int K,
    int lda, int ldb, int ldc,
    const float* __restrict__ bias, int relu)
{
    int lane = threadIdx.x & 31;
    int wave = threadIdx.x >> 5;
    int tilesN = N >> 4;
    int tiles  = (M >> 4) * tilesN;
    int tile   = blockIdx.x * 8 + wave;
    if (tile >= tiles) return;                    // uniform per wave
    int tm = tile / tilesN, tn = tile % tilesN;
    const _Float16* Abase = A  + (size_t)(tm * 16) * lda;
    const _Float16* Bbase = Bm + (size_t)(tn * 16) * ldb;
    v8f c = {};
    for (int kk = 0; kk < K; kk += 32) {
        v16h a = load_fragA(Abase, lda, kk, lane);
        v16h b = load_fragB(Bbase, ldb, kk, lane);
        c = __builtin_amdgcn_wmma_f32_16x16x32_f16(false, a, false, b, (short)0, c,
                                                   false, false);
    }
    int l15 = lane & 15, lg = lane >> 4;
    int col = tn * 16 + l15;
    float bv = bias ? bias[col] : 0.0f;
#pragma unroll
    for (int v = 0; v < 8; ++v) {
        int row = tm * 16 + lg * 8 + v;
        float val = c[v] + bv;
        if (relu) val = fmaxf(val, 0.0f);
        C[(size_t)row * ldc + col] = val;
    }
}

// ---------------- Char CNN: conv1d(k=3,SAME) + relu + maxpool ---------------
__global__ void char_cnn_kernel(const float* __restrict__ cemb,
                                const float* __restrict__ cw,   // (CH,CD,3)
                                const float* __restrict__ cb,
                                const int*   __restrict__ words,
                                float* __restrict__ tok)        // (NW+1, CH)
{
    int w = blockIdx.x;
    int tid = threadIdx.x;
    __shared__ float ct[CL_][CD_];
    for (int i = tid; i < CL_ * CD_; i += 32) {
        int p = i / CD_, d = i % CD_;
        ct[p][d] = cemb[(size_t)words[w * CL_ + p] * CD_ + d];
    }
    __syncthreads();
    if (tid < CH_) {
        int ch = tid;
        float m = -1e30f;
        for (int pos = 0; pos < CL_; ++pos) {
            float acc = cb[ch];
            for (int k = 0; k < 3; ++k) {
                int p = pos + k - 1;
                if (p >= 0 && p < CL_)
                    for (int d = 0; d < CD_; ++d)
                        acc += ct[p][d] * cw[(ch * CD_ + d) * 3 + k];
            }
            m = fmaxf(m, fmaxf(acc, 0.0f));
        }
        tok[(size_t)(w + 1) * CH_ + ch] = m;
    }
    if (w == 0 && tid < CH_) tok[tid] = 0.0f;    // prepended zero row
}

// ---------------- texts_f16 = [word_emb | tok_enc | zero-pad] ---------------
__global__ void build_texts_kernel(const float* __restrict__ wemb,
                                   const float* __restrict__ tok,
                                   const int*   __restrict__ tokens,
                                   const int*   __restrict__ widx,
                                   _Float16* __restrict__ texts)
{
    int gid = blockIdx.x * blockDim.x + threadIdx.x;
    int total = B_ * S_ * K0P;
    if (gid >= total) return;
    int row = gid / K0P, col = gid % K0P;
    float v = 0.0f;
    if (col < D_)            v = wemb[(size_t)tokens[row] * D_ + col];
    else if (col < D_ + CH_) v = tok[(size_t)widx[row] * CH_ + (col - D_)];
    texts[gid] = (_Float16)v;
}

// ---------------- f32 -> f16 weight pack (K padded, optional transpose) -----
__global__ void pack_f16_kernel(const float* __restrict__ src,
                                _Float16* __restrict__ dst,
                                int R, int K, int Kp, int transpose)
{
    int gid = blockIdx.x * blockDim.x + threadIdx.x;
    if (gid >= R * Kp) return;
    int r = gid / Kp, k = gid % Kp;
    float v = 0.0f;
    if (k < K) v = transpose ? src[(size_t)k * R + r] : src[(size_t)r * K + k];
    dst[gid] = (_Float16)v;
}

// ---------------- GRU scan ---------------------------------------------------
// Grid: 8 blocks = (dir in {0,1}) x (4 chunks of 16 batch rows).
// Each block: 256 threads = 8 waves; per step M=16, N=384, K=128 -> 24 WMMA
// tiles (3 per wave); hg kept entirely in LDS; b_ih pre-folded into xg,
// b_hh folded into the hg tile epilogue.
__global__ void __launch_bounds__(256) gru_scan_kernel(
    const float*    __restrict__ xg,    // [B,S,768]  x-gates (b_ih included)
    const _Float16* __restrict__ whh,   // [2,384,128] f16
    const float*    __restrict__ bhh,   // [2,384]
    const int*      __restrict__ lens,  // [B]
    _Float16*       __restrict__ hcat)  // [B,S,256] (dir0 -> 0..127, dir1 -> 128..255)
{
    int dir   = blockIdx.x & 1;
    int chunk = blockIdx.x >> 1;
    int b0    = chunk * BCH;
    int tid   = threadIdx.x;
    int lane  = tid & 31, wave = tid >> 5;
    __shared__ __align__(16) float    hgs[BCH][G3_];   // 24 KB
    __shared__ __align__(16) float    hsf[BCH][H_];    //  8 KB
    __shared__ __align__(16) _Float16 hsh[BCH][H_];    //  4 KB
    for (int i = tid; i < BCH * H_; i += 256) {
        (&hsf[0][0])[i] = 0.0f;
        (&hsh[0][0])[i] = (_Float16)0.0f;
    }
    __syncthreads();
    const _Float16* whhd = whh + (size_t)dir * G3_ * H_;
    const float*    bh   = bhh + dir * G3_;
    for (int step = 0; step < S_; ++step) {
        int s = dir ? (S_ - 1 - step) : step;
        // hg = h @ whh^T + b_hh : M=16, N=384, K=128 -> 24 tiles over 8 waves
        for (int t = wave; t < G3_ / 16; t += 8) {
            int tn = t;
            v8f c = {};
            const _Float16* Abase = &hsh[0][0];
            const _Float16* Bbase = whhd + (size_t)(tn * 16) * H_;
#pragma unroll
            for (int kk = 0; kk < H_; kk += 32) {
                v16h a = load_fragA(Abase, H_, kk, lane);
                v16h b = load_fragB(Bbase, H_, kk, lane);
                c = __builtin_amdgcn_wmma_f32_16x16x32_f16(false, a, false, b, (short)0,
                                                           c, false, false);
            }
            int l15 = lane & 15, lg = lane >> 4;
            int col = tn * 16 + l15;
            float bv = bh[col];
#pragma unroll
            for (int v = 0; v < 8; ++v)
                hgs[lg * 8 + v][col] = c[v] + bv;
        }
        __syncthreads();
        // elementwise gates: 2048 elems / 256 threads = 8 each (no divergence)
#pragma unroll
        for (int e = 0; e < (BCH * H_) / 256; ++e) {
            int idx = e * 256 + tid;
            int bl = idx >> 7, u = idx & 127;
            int b  = b0 + bl;
            const float* xgp = xg + (size_t)(b * S_ + s) * N01 + dir * G3_;
            float xr = xgp[u];
            float xz = xgp[H_ + u];
            float xn = xgp[2 * H_ + u];
            float hr = hgs[bl][u];
            float hz = hgs[bl][H_ + u];
            float hn = hgs[bl][2 * H_ + u];
            float r = 1.0f / (1.0f + __expf(-(xr + hr)));
            float z = 1.0f / (1.0f + __expf(-(xz + hz)));
            float n = tanhf(xn + r * hn);
            float hprev = hsf[bl][u];
            float hnew  = (1.0f - z) * n + z * hprev;
            float hcur  = (s < lens[b]) ? hnew : hprev;
            hsf[bl][u] = hcur;
            hsh[bl][u] = (_Float16)hcur;
            hcat[(size_t)(b * S_ + s) * (2 * H_) + dir * H_ + u] = (_Float16)hcur;
        }
        __syncthreads();
    }
}

// ---------------- MLP2: emissions = hmlp @ w2 + b2  (N=17, scalar) ----------
__global__ void mlp2_kernel(const float* __restrict__ hm, const float* __restrict__ w2,
                            const float* __restrict__ b2, float* __restrict__ em)
{
    int gid = blockIdx.x * blockDim.x + threadIdx.x;
    if (gid >= B_ * S_ * T_) return;
    int m = gid / T_, t = gid % T_;
    float acc = b2[t];
    for (int k = 0; k < H_; ++k) acc += hm[(size_t)m * H_ + k] * w2[k * T_ + t];
    em[gid] = acc;
}

// ---------------- CRF gold-path score ---------------------------------------
__global__ void crf_score_kernel(const float* __restrict__ em, const int* __restrict__ tg,
                                 const int* __restrict__ lens,
                                 const float* __restrict__ cs, const float* __restrict__ ce,
                                 const float* __restrict__ tr, float* __restrict__ score)
{
    int b = threadIdx.x;
    if (b >= B_) return;
    int len = lens[b];
    const int*   tgb = tg + b * S_;
    const float* emb = em + (size_t)b * S_ * T_;
    float sc = cs[tgb[0]] + emb[tgb[0]];
    for (int s = 1; s < S_; ++s)
        if (s < len) sc += tr[tgb[s - 1] * T_ + tgb[s]] + emb[s * T_ + tgb[s]];
    sc += ce[tgb[len - 1]];
    score[b] = sc;
}

// ---------------- CRF forward algorithm (log partition) ---------------------
__global__ void crf_forward_kernel(const float* __restrict__ em, const int* __restrict__ lens,
                                   const float* __restrict__ cs, const float* __restrict__ ce,
                                   const float* __restrict__ tr,
                                   const float* __restrict__ score, float* __restrict__ llh)
{
    int b = blockIdx.x, tid = threadIdx.x;
    __shared__ float trs[T_ * T_];
    __shared__ float alpha[T_], na[T_];
    for (int i = tid; i < T_ * T_; i += 32) trs[i] = tr[i];
    const float* emb = em + (size_t)b * S_ * T_;
    if (tid < T_) alpha[tid] = cs[tid] + emb[tid];
    __syncthreads();
    int len = lens[b];
    for (int s = 1; s < S_; ++s) {
        if (tid < T_) {
            float mx = -1e30f;
            for (int i = 0; i < T_; ++i) mx = fmaxf(mx, alpha[i] + trs[i * T_ + tid]);
            float sum = 0.0f;
            for (int i = 0; i < T_; ++i) sum += __expf(alpha[i] + trs[i * T_ + tid] - mx);
            float nj = mx + __logf(sum) + emb[s * T_ + tid];
            na[tid] = (s < len) ? nj : alpha[tid];
        }
        __syncthreads();
        if (tid < T_) alpha[tid] = na[tid];
        __syncthreads();
    }
    if (tid == 0) {
        float mx = -1e30f;
        for (int j = 0; j < T_; ++j) mx = fmaxf(mx, alpha[j] + ce[j]);
        float sum = 0.0f;
        for (int j = 0; j < T_; ++j) sum += __expf(alpha[j] + ce[j] - mx);
        llh[b] = score[b] - (mx + __logf(sum));
    }
}

// ---------------- final loss reduction --------------------------------------
__global__ void loss_kernel(const float* __restrict__ llh, const int* __restrict__ lens,
                            float* __restrict__ out)
{
    if (threadIdx.x == 0) {
        float s = 0.0f, msum = 0.0f;
        for (int b = 0; b < B_; ++b) { s += llh[b]; msum += (float)lens[b]; }
        out[0] = -(s / msum);
    }
}

extern "C" void kernel_launch(void* const* d_in, const int* in_sizes, int n_in,
                              void* d_out, int out_size, void* d_ws, size_t ws_size,
                              hipStream_t stream) {
    (void)in_sizes; (void)n_in; (void)out_size; (void)ws_size;
    const float* char_emb = (const float*)d_in[0];
    const float* conv_w   = (const float*)d_in[1];
    const float* conv_b   = (const float*)d_in[2];
    const float* word_emb = (const float*)d_in[3];
    const float* w_ih_l0  = (const float*)d_in[4];
    const float* w_hh_l0  = (const float*)d_in[5];
    const float* b_ih_l0  = (const float*)d_in[6];
    const float* b_hh_l0  = (const float*)d_in[7];
    const float* w_ih_l1  = (const float*)d_in[8];
    const float* w_hh_l1  = (const float*)d_in[9];
    const float* b_ih_l1  = (const float*)d_in[10];
    const float* b_hh_l1  = (const float*)d_in[11];
    const float* mlp_w1   = (const float*)d_in[12];
    const float* mlp_b1   = (const float*)d_in[13];
    const float* mlp_w2   = (const float*)d_in[14];
    const float* mlp_b2   = (const float*)d_in[15];
    const float* crf_start= (const float*)d_in[16];
    const float* crf_end  = (const float*)d_in[17];
    const float* crf_trans= (const float*)d_in[18];
    const int* bwords     = (const int*)d_in[19];
    // d_in[20] = batched_char_words_len (unused by forward)
    const int* widx       = (const int*)d_in[21];
    const int* tokens     = (const int*)d_in[22];
    const int* lens       = (const int*)d_in[23];
    const int* target     = (const int*)d_in[24];
    float* out = (float*)d_out;

    char* ws = (char*)d_ws;
    size_t off = 0;
    auto alloc = [&](size_t bytes) -> void* {
        void* p = ws + off;
        off = (off + bytes + 255) & ~(size_t)255;
        return p;
    };
    float*    tok_enc = (float*)   alloc((size_t)(NW_ + 1) * CH_ * 4);
    _Float16* texts16 = (_Float16*)alloc((size_t)B_ * S_ * K0P * 2);
    _Float16* wih0    = (_Float16*)alloc((size_t)N01 * K0P * 2);
    _Float16* wih1    = (_Float16*)alloc((size_t)N01 * K1P * 2);
    _Float16* whh0    = (_Float16*)alloc((size_t)N01 * H_ * 2);
    _Float16* whh1    = (_Float16*)alloc((size_t)N01 * H_ * 2);
    _Float16* w1t     = (_Float16*)alloc((size_t)H_ * (2 * H_) * 2);
    float*    xg      = (float*)   alloc((size_t)B_ * S_ * N01 * 4);   // reused l0/l1
    _Float16* hcat0   = (_Float16*)alloc((size_t)B_ * S_ * 2 * H_ * 2);
    _Float16* hcat1   = (_Float16*)alloc((size_t)B_ * S_ * 2 * H_ * 2);
    float*    hmlp    = (float*)   alloc((size_t)B_ * S_ * H_ * 4);
    float*    emis    = (float*)   alloc((size_t)B_ * S_ * T_ * 4);
    float*    score   = (float*)   alloc((size_t)B_ * 4);
    float*    llh     = (float*)   alloc((size_t)B_ * 4);

    // 1) char CNN -> tok_enc (padded with zero row 0)
    char_cnn_kernel<<<NW_, 32, 0, stream>>>(char_emb, conv_w, conv_b, bwords, tok_enc);
    // 2) weight packs to f16
    pack_f16_kernel<<<(N01 * K0P + 255) / 256, 256, 0, stream>>>(w_ih_l0, wih0, N01, 330, K0P, 0);
    pack_f16_kernel<<<(N01 * K1P + 255) / 256, 256, 0, stream>>>(w_ih_l1, wih1, N01, 256, K1P, 0);
    pack_f16_kernel<<<(N01 * H_  + 255) / 256, 256, 0, stream>>>(w_hh_l0, whh0, N01, H_, H_, 0);
    pack_f16_kernel<<<(N01 * H_  + 255) / 256, 256, 0, stream>>>(w_hh_l1, whh1, N01, H_, H_, 0);
    pack_f16_kernel<<<(H_ * 2 * H_ + 255) / 256, 256, 0, stream>>>(mlp_w1, w1t, H_, 2 * H_, 2 * H_, 1);
    // 3) texts_f16
    build_texts_kernel<<<(B_ * S_ * K0P + 255) / 256, 256, 0, stream>>>(
        word_emb, tok_enc, tokens, widx, texts16);
    // 4) layer-0 input gates: (16384,352)x(352,768), b_ih folded in
    {
        int tiles = (B_ * S_ / 16) * (N01 / 16);
        wmma_gemm_kernel<<<(tiles + 7) / 8, 256, 0, stream>>>(
            texts16, wih0, xg, B_ * S_, N01, K0P, K0P, K0P, N01, b_ih_l0, 0);
    }
    // 5) GRU layer 0 scan: 8 blocks = 2 dirs x 4 batch chunks
    gru_scan_kernel<<<8, 256, 0, stream>>>(xg, whh0, b_hh_l0, lens, hcat0);
    // 6) layer-1 input gates: (16384,256)x(256,768), b_ih folded in
    {
        int tiles = (B_ * S_ / 16) * (N01 / 16);
        wmma_gemm_kernel<<<(tiles + 7) / 8, 256, 0, stream>>>(
            hcat0, wih1, xg, B_ * S_, N01, K1P, K1P, K1P, N01, b_ih_l1, 0);
    }
    // 7) GRU layer 1 scan
    gru_scan_kernel<<<8, 256, 0, stream>>>(xg, whh1, b_hh_l1, lens, hcat1);
    // 8) MLP1: (16384,256)x(256,128) + bias + relu
    {
        int tiles = (B_ * S_ / 16) * (H_ / 16);
        wmma_gemm_kernel<<<(tiles + 7) / 8, 256, 0, stream>>>(
            hcat1, w1t, hmlp, B_ * S_, H_, 2 * H_, 2 * H_, 2 * H_, H_, mlp_b1, 1);
    }
    // 9) MLP2 -> emissions
    mlp2_kernel<<<(B_ * S_ * T_ + 255) / 256, 256, 0, stream>>>(hmlp, mlp_w2, mlp_b2, emis);
    // 10) CRF
    crf_score_kernel<<<1, 64, 0, stream>>>(emis, target, lens, crf_start, crf_end, crf_trans, score);
    crf_forward_kernel<<<B_, 32, 0, stream>>>(emis, lens, crf_start, crf_end, crf_trans, score, llh);
    loss_kernel<<<1, 32, 0, stream>>>(llh, lens, out);
}
